// MultiheadAssignmentAttention_56693568307507
// MI455X (gfx1250) — matrix operand order repaired and checked by reading, not compiled, not run on hardware
//
#include <hip/hip_runtime.h>
#include <hip/hip_bf16.h>

// ---------------------------------------------------------------------------
// MultiheadAssignmentAttention for MI455X (gfx1250, wave32, WMMA bf16,
// async global->LDS staging via CDNA5 GLOBAL_LOAD_ASYNC_TO_LDS + ASYNCcnt)
// ---------------------------------------------------------------------------

typedef __bf16 bf16_t;
typedef __attribute__((ext_vector_type(16))) __bf16 v16bf;
typedef __attribute__((ext_vector_type(8)))  __bf16 v8bf;
typedef __attribute__((ext_vector_type(8)))  float   v8f;

#define QLEN   64
#define NLEN   1024
#define BSZ    64
#define DMODEL 256
#define NHEAD  8
#define DHEAD  32
#define NC     64   // n-chunk width in attention kernel

__device__ __forceinline__ v8f wmma_bf16(v16bf a, v16bf b, v8f c) {
  // v_wmma_f32_16x16x32_bf16  D = A(16x32) * B(32x16) + C
  return __builtin_amdgcn_wmma_f32_16x16x32_bf16(
      /*neg_a=*/false, a, /*neg_b=*/false, b,
      /*c_mod=*/(short)0, c, /*reuse_a=*/false, /*reuse_b=*/false);
}

// CDNA5 async global->LDS copy (16B per lane), tracked by ASYNCcnt.
// vdst = LDS byte address (workgroup-relative), vaddr = 64-bit global addr.
__device__ __forceinline__ void async_load_b128(unsigned lds_off, const void* gaddr) {
  asm volatile("global_load_async_to_lds_b128 %0, %1, off"
               :: "v"(lds_off), "v"(gaddr) : "memory");
}
__device__ __forceinline__ void wait_async0() {
  asm volatile("s_wait_asynccnt 0x0" ::: "memory");
}
// Generic LDS pointer -> 32-bit LDS offset (low 32 bits of the flat address).
__device__ __forceinline__ unsigned lds_off32(const void* p) {
  return (unsigned)(uintptr_t)p;
}

// A-fragment (16x32 bf16): lane l holds row M=l%16; with g=l/16,
// element e -> K=g*16+e. One contiguous 16-run (loaded as two 8-runs).
__device__ __forceinline__ v16bf load_a_bf16(const bf16_t* row, int g) {
  v8bf lo = *(const v8bf*)(row + g * 16);
  v8bf hi = *(const v8bf*)(row + g * 16 + 8);
  v16bf a;
#pragma unroll
  for (int i = 0; i < 8; i++) { a[i] = lo[i]; a[8 + i] = hi[i]; }
  return a;
}

// A-fragment built from an fp32 row (on-the-fly bf16 conversion).
__device__ __forceinline__ v16bf load_a_f32(const float* row, int g) {
  v16bf a;
#pragma unroll
  for (int i = 0; i < 8; i++) {
    a[i]     = (bf16_t)row[g * 16 + i];
    a[8 + i] = (bf16_t)row[g * 16 + 8 + i];
  }
  return a;
}

// ---------------------------------------------------------------------------
// fp32 -> bf16 elementwise convert (for weight matrices)
// ---------------------------------------------------------------------------
__global__ void cvt_bf16_kernel(const float* __restrict__ src,
                                bf16_t* __restrict__ dst, int n) {
  int i = blockIdx.x * blockDim.x + threadIdx.x;
  if (i < n) dst[i] = (bf16_t)src[i];
}

// ---------------------------------------------------------------------------
// Generic 256-wide GEMM:  C[m, j] = sum_i A[m,i] * W[j,i]  + bias[j]
//   - one 16x16 output tile per wave, K=256 in 8 WMMA steps
//   - B-fragment reads W row-major directly (per-lane contiguous K)
// MODE 0: A=query fp32 (m = q*64+b) -> qp  bf16 [b][h][q][dh]
// MODE 1: A=key   fp32 (m = n*64+b) -> kp  bf16 [b][h][n][dh]
// MODE 2: A=value fp32 (m = n*64+b) -> vt  bf16 [b][h][dh][n]   (transposed)
// MODE 3: A=num  bf16  (m = b*64+q) -> out fp32 [q][b][j]       (d_out)
// ---------------------------------------------------------------------------
template <int MODE>
__global__ void gemm256_kernel(const void* __restrict__ A_,
                               const bf16_t* __restrict__ Wb,
                               const float* __restrict__ bias,
                               void* __restrict__ out_) {
  const int wave = threadIdx.x >> 5;
  const int lane = threadIdx.x & 31;
  const int g    = lane >> 4;
  const int ln   = lane & 15;
  const int tile = blockIdx.x * 8 + wave;
  const int rowTile = tile >> 4;   // 16 column tiles (256/16)
  const int colTile = tile & 15;

  const int mA = rowTile * 16 + ln;                        // A row for this lane
  const bf16_t* wrow = Wb + (size_t)(colTile * 16 + ln) * DMODEL;  // N = ln

  v8f acc = {};
#pragma unroll
  for (int k0 = 0; k0 < DMODEL; k0 += 32) {
    v16bf a;
    if (MODE == 3) {
      a = load_a_bf16((const bf16_t*)A_ + (size_t)mA * DMODEL + k0, g);
    } else {
      a = load_a_f32((const float*)A_ + (size_t)mA * DMODEL + k0, g);
    }
    // B-fragment: lane group g holds K = k0 + g*16 + e (16 contiguous bf16)
    v16bf b = *(const v16bf*)(wrow + k0 + g * 16);
    acc = wmma_bf16(a, b, acc);
  }

  const int   jn = colTile * 16 + ln;   // output feature
  const float bj = bias[jn];
  const int   h  = jn >> 5;             // head (DHEAD = 32)
  const int   dh = jn & 31;

#pragma unroll
  for (int r = 0; r < 8; r++) {
    const int   m = rowTile * 16 + g * 8 + r;   // C layout: M = g*8 + r
    const float v = acc[r] + bj;
    if (MODE == 0) {
      int qpos = m / BSZ, b = m % BSZ;
      ((bf16_t*)out_)[(((size_t)(b * NHEAD + h) * QLEN + qpos) * DHEAD + dh)] = (bf16_t)v;
    } else if (MODE == 1) {
      int n = m / BSZ, b = m % BSZ;
      ((bf16_t*)out_)[(((size_t)(b * NHEAD + h) * NLEN + n) * DHEAD + dh)] = (bf16_t)v;
    } else if (MODE == 2) {
      int n = m / BSZ, b = m % BSZ;
      ((bf16_t*)out_)[(((size_t)(b * NHEAD + h) * DHEAD + dh) * NLEN + n)] = (bf16_t)v;
    } else {
      int b = m / QLEN, q = m % QLEN;
      ((float*)out_)[((size_t)q * BSZ + b) * DMODEL + jn] = v;
    }
  }
}

// ---------------------------------------------------------------------------
// Attention core. One block per batch element b; 8 waves / 256 threads.
// Per 64-wide n-chunk, loop over the 8 heads:
//   k/v chunks (4KB each) staged into LDS with GLOBAL_LOAD_ASYNC_TO_LDS_B128
//   scores(64xNC) = q(64x32) @ k^T   (WMMA, dh=32 = one K-step)
//   softmax over q per column (4-way parallel reduce), mask, head means in LDS
//   num(64x32)  += P(64xNC) @ v(NCx32)  (WMMA, C-fragments persist in VGPRs)
// Means are written directly (no atomics -> deterministic).
// ---------------------------------------------------------------------------
__global__ void attn_kernel(const bf16_t* __restrict__ qp,
                            const bf16_t* __restrict__ kp,
                            const bf16_t* __restrict__ vt,
                            const unsigned char* __restrict__ mask,
                            bf16_t* __restrict__ num_bf,
                            float* __restrict__ P_mean,
                            float* __restrict__ S_mean) {
  __shared__ float sS[QLEN * NC];    // scores, then P, for current head
  __shared__ float sPm[QLEN * NC];   // sum over heads of P
  __shared__ float sSm[QLEN * NC];   // sum over heads of masked scores
  __shared__ float sMax[NC];
  __shared__ float sInv[NC];
  __shared__ float sRed[4 * NC];     // softmax partial reductions
  __shared__ alignas(32) bf16_t sK[NC * DHEAD];   // [n][dh]  4KB, async-staged
  __shared__ alignas(32) bf16_t sV[DHEAD * NC];   // [dh][n]  4KB, async-staged

  const int b    = blockIdx.x;
  const int t    = threadIdx.x;
  const int wave = t >> 5;
  const int lane = t & 31;
  const int g    = lane >> 4;
  const int ln   = lane & 15;
  const int qt   = wave >> 1;        // this wave's q-tile for num GEMM (0..3)
  const int dt   = wave & 1;         // this wave's dh-tile for num GEMM (0..1)

  const float inv_sqrt_dh = 0.17677669529663687f;  // 1/sqrt(32)

  const unsigned sK_base = lds_off32(sK);
  const unsigned sV_base = lds_off32(sV);

  v8f accN[NHEAD];
#pragma unroll
  for (int h = 0; h < NHEAD; h++) accN[h] = (v8f){};

  for (int n0 = 0; n0 < NLEN; n0 += NC) {
    for (int i = t; i < QLEN * NC; i += 256) { sPm[i] = 0.f; sSm[i] = 0.f; }
    __syncthreads();

#pragma unroll
    for (int h = 0; h < NHEAD; h++) {
      const bf16_t* qbase = qp + ((size_t)(b * NHEAD + h) * QLEN) * DHEAD;

      // ---- async-stage k chunk [NC][DHEAD] and v chunk [DHEAD][NC] into LDS
      {
        // sK: 256 threads x 16B = 4KB; row n = t/4, 16B segment t%4
        const bf16_t* kg = kp + ((size_t)(b * NHEAD + h) * NLEN + n0 + (t >> 2)) * DHEAD
                              + (t & 3) * 8;
        async_load_b128(sK_base + (unsigned)t * 16u, kg);
        // sV: row dh = t/8 (64 elems = 128B per row), 16B segment t%8
        const bf16_t* vg = vt + ((size_t)(b * NHEAD + h) * DHEAD + (t >> 3)) * NLEN
                              + n0 + (t & 7) * 8;
        async_load_b128(sV_base + (unsigned)t * 16u, vg);
        wait_async0();
      }
      __syncthreads();

      // ---- scores: 16 tiles of 16x16, 2 per wave, one WMMA each (K=dh=32)
      for (int tt = wave; tt < 16; tt += 8) {
        const int sqt = tt >> 2, snt = tt & 3;
        v16bf a  = load_a_bf16(qbase + (size_t)(sqt * 16 + ln) * DHEAD, g);
        // B: N = n (lane), per-lane contiguous dh from LDS-staged k
        v16bf bb = *(const v16bf*)(sK + (snt * 16 + ln) * DHEAD + g * 16);
        v8f c = {};
        c = wmma_bf16(a, bb, c);
#pragma unroll
        for (int r = 0; r < 8; r++) {
          const int qq = sqt * 16 + g * 8 + r;
          const int nn = snt * 16 + ln;
          sS[qq * NC + nn] = c[r] * inv_sqrt_dh;
        }
      }
      __syncthreads();

      // ---- softmax over q (64 values) per column n, 4-way parallel
      {
        const int col = t & 63, qg = t >> 6;      // qg in 0..3 -> 16 q each
        float mx = -1e30f;
        for (int q = qg * 16; q < qg * 16 + 16; q++) mx = fmaxf(mx, sS[q * NC + col]);
        sRed[qg * NC + col] = mx;
      }
      __syncthreads();
      if (t < NC) {
        sMax[t] = fmaxf(fmaxf(sRed[t], sRed[NC + t]),
                        fmaxf(sRed[2 * NC + t], sRed[3 * NC + t]));
      }
      __syncthreads();
      {
        const int col = t & 63, qg = t >> 6;
        float sum = 0.f;
        const float mx = sMax[col];
        for (int q = qg * 16; q < qg * 16 + 16; q++) sum += __expf(sS[q * NC + col] - mx);
        sRed[qg * NC + col] = sum;
      }
      __syncthreads();
      if (t < NC) {
        sInv[t] = 1.0f / (sRed[t] + sRed[NC + t] + sRed[2 * NC + t] + sRed[3 * NC + t]);
      }
      __syncthreads();

      // ---- P = exp(s-max)*inv ; mask whole key-columns ; accumulate means
      for (int i = t; i < QLEN * NC; i += 256) {
        const int nn = i & (NC - 1);
        float sv = sS[i];
        float p  = __expf(sv - sMax[nn]) * sInv[nn];
        if (mask[(size_t)b * NLEN + n0 + nn] != 0) { p = 0.f; sv = 0.f; }
        sS[i]   = p;      // P for the num GEMM
        sPm[i] += p;
        sSm[i] += sv;
      }
      __syncthreads();

      // ---- num += P(64xNC) @ v(NCx32); wave owns (qt, dt) 16x16 tile
#pragma unroll
      for (int ks = 0; ks < NC / 32; ks++) {
        const float* prow = &sS[(qt * 16 + ln) * NC + ks * 32];
        v16bf a;
#pragma unroll
        for (int i = 0; i < 8; i++) {
          a[i]     = (bf16_t)prow[g * 16 + i];
          a[8 + i] = (bf16_t)prow[g * 16 + 8 + i];
        }
        // B: N = dh (lane), per-lane contiguous n from LDS-staged v (transposed)
        v16bf bb = *(const v16bf*)(sV + (dt * 16 + ln) * NC + ks * 32 + g * 16);
        accN[h] = wmma_bf16(a, bb, accN[h]);
      }
      __syncthreads();  // sS / sK / sV reused by next head
    }

    // ---- write head-means for this chunk (plain stores, deterministic)
    for (int i = t; i < QLEN * NC; i += 256) {
      const int q = i / NC, nn = i & (NC - 1);
      const size_t idx = ((size_t)b * QLEN + q) * NLEN + n0 + nn;
      P_mean[idx] = sPm[i] * 0.125f;
      S_mean[idx] = sSm[i] * 0.125f;
    }
    __syncthreads();
  }

  // ---- spill num accumulators: num_bf[b][q][h*32 + dh]  (bf16 for out-proj)
#pragma unroll
  for (int h = 0; h < NHEAD; h++) {
#pragma unroll
    for (int r = 0; r < 8; r++) {
      const int q    = qt * 16 + g * 8 + r;
      const int feat = h * DHEAD + dt * 16 + ln;
      num_bf[((size_t)b * QLEN + q) * DMODEL + feat] = (bf16_t)accN[h][r];
    }
  }
}

// ---------------------------------------------------------------------------
// Host-side orchestration
// ---------------------------------------------------------------------------
extern "C" void kernel_launch(void* const* d_in, const int* in_sizes, int n_in,
                              void* d_out, int out_size, void* d_ws, size_t ws_size,
                              hipStream_t stream) {
  (void)in_sizes; (void)n_in; (void)out_size; (void)ws_size;

  const float* query = (const float*)d_in[0];
  const float* key   = (const float*)d_in[1];
  const float* value = (const float*)d_in[2];
  const unsigned char* mask = (const unsigned char*)d_in[3];
  const float* Wq = (const float*)d_in[4];  const float* bq = (const float*)d_in[5];
  const float* Wk = (const float*)d_in[6];  const float* bk = (const float*)d_in[7];
  const float* Wv = (const float*)d_in[8];  const float* bv = (const float*)d_in[9];
  const float* Wo = (const float*)d_in[10]; const float* bo = (const float*)d_in[11];

  float* out    = (float*)d_out;                               // [Q][B][256]
  float* P_mean = out + (size_t)QLEN * BSZ * DMODEL;           // [B][Q][N]
  float* S_mean = P_mean + (size_t)BSZ * QLEN * NLEN;          // [B][Q][N]

  // scratch layout (all offsets 256B-aligned for 32B vector loads)
  char* ws = (char*)d_ws;
  bf16_t* Wq_b   = (bf16_t*)(ws + 0);
  bf16_t* Wk_b   = (bf16_t*)(ws + 131072);
  bf16_t* Wv_b   = (bf16_t*)(ws + 262144);
  bf16_t* Wo_b   = (bf16_t*)(ws + 393216);
  bf16_t* qp     = (bf16_t*)(ws + 524288);     //  2 MB  [b][h][q][dh]
  bf16_t* kp     = (bf16_t*)(ws + 2621440);    // 32 MB  [b][h][n][dh]
  bf16_t* vt     = (bf16_t*)(ws + 36175872);   // 32 MB  [b][h][dh][n]
  bf16_t* num_bf = (bf16_t*)(ws + 69730304);   //  2 MB  [b][q][256]

  // 1) weights -> bf16
  const int wN = DMODEL * DMODEL;
  cvt_bf16_kernel<<<(wN + 255) / 256, 256, 0, stream>>>(Wq, Wq_b, wN);
  cvt_bf16_kernel<<<(wN + 255) / 256, 256, 0, stream>>>(Wk, Wk_b, wN);
  cvt_bf16_kernel<<<(wN + 255) / 256, 256, 0, stream>>>(Wv, Wv_b, wN);
  cvt_bf16_kernel<<<(wN + 255) / 256, 256, 0, stream>>>(Wo, Wo_b, wN);

  // 2) projections: rows/16 * 16 col-tiles / 8 waves per block
  gemm256_kernel<0><<<(QLEN * BSZ / 16) * 16 / 8, 256, 0, stream>>>(query, Wq_b, bq, qp);
  gemm256_kernel<1><<<(NLEN * BSZ / 16) * 16 / 8, 256, 0, stream>>>(key,   Wk_b, bk, kp);
  gemm256_kernel<2><<<(NLEN * BSZ / 16) * 16 / 8, 256, 0, stream>>>(value, Wv_b, bv, vt);

  // 3) attention core: one block per batch element
  attn_kernel<<<BSZ, 256, 0, stream>>>(qp, kp, vt, mask, num_bf, P_mean, S_mean);

  // 4) output projection -> d_out
  gemm256_kernel<3><<<(BSZ * QLEN / 16) * 16 / 8, 256, 0, stream>>>(num_bf, Wo_b, bo, out);
}
